// SAGCNNet_89129161327119
// MI455X (gfx1250) — matrix operand order repaired and checked by reading, not compiled, not run on hardware
//
#include <hip/hip_runtime.h>
#include <hip/hip_bf16.h>
#include <math.h>

// ---------------------------------------------------------------------------
// Types for CDNA5 WMMA
// ---------------------------------------------------------------------------
typedef __attribute__((ext_vector_type(16))) _Float16 v16h;
typedef __attribute__((ext_vector_type(8)))  float    v8f;
typedef __attribute__((ext_vector_type(2)))  _Float16 h2_t;
typedef __attribute__((ext_vector_type(4)))  _Float16 h4_t;
typedef __attribute__((ext_vector_type(4)))  float    f4_t;

#define HDIM 256
#define GEMM_BLK 128   // 4 wave32s
#define TILE_M 128     // block tile rows (wave tile = 32 rows)
#define TILE_N 64
#define LDS_STRIDE 36  // halves per row: 72B -> 8B aligned, conflict-free

// ---------------------------------------------------------------------------
// Tiled WMMA GEMM:  C[M,N] = act( A[M,K](f32) * B(f32) + bias [+ resid] )
//   TRANSB=0: B row-major [K,N] (k*ldb+n);  TRANSB=1: B row-major [N,K] -> A@B^T
// Block computes 128x64 C tile. Wave w owns rows 32w..32w+31 (2 A fragments),
// sweeps four 16-col subtiles; each B fragment gather feeds 2 WMMAs.
// A/B converted to f16 in LDS; f32 accumulate via v_wmma_f32_16x16x32_f16.
// Fast path (interior tile): float4 global loads + packed ds_store_b64.
// ---------------------------------------------------------------------------
template <int ACT, bool TRANSB, bool RESID>
__global__ __launch_bounds__(GEMM_BLK)
void gemm_wmma(const float* __restrict__ A, const float* __restrict__ Bw,
               const float* __restrict__ bias, const float* __restrict__ resid,
               float* __restrict__ C, int M, int Nc, int K,
               int lda, int ldb, int ldc) {
  __shared__ _Float16 As[TILE_M][LDS_STRIDE];
  __shared__ _Float16 Bs[TILE_N][LDS_STRIDE];   // stored [n][k]

  const int tid  = threadIdx.x;
  const int wave = tid >> 5;
  const int lane = tid & 31;
  const int hi   = lane >> 4;   // half-wave select (ISA 16x16 layouts)
  const int ln   = lane & 15;

  const int m0 = blockIdx.x * TILE_M;
  const int n0 = blockIdx.y * TILE_N;

  const bool fastA = (m0 + TILE_M <= M) && ((lda & 3) == 0);
  const bool fastB = (n0 + TILE_N <= Nc) && (!TRANSB || ((ldb & 3) == 0));

  v8f acc[2][4] = {};

  for (int k0 = 0; k0 < K; k0 += 32) {
    const bool fullK = (k0 + 32 <= K);

    // ---- stage A tile (128 x 32) as f16 ----
    if (fastA && fullK) {
#pragma unroll
      for (int q = 0; q < 8; ++q) {            // 128*32/4 = 1024 float4 chunks
        int idx4 = tid + q * GEMM_BLK;
        int r = idx4 >> 3;                     // 8 chunks per 32-wide row
        int c = (idx4 & 7) * 4;
        f4_t v = *(const f4_t*)&A[(size_t)(m0 + r) * lda + (k0 + c)];
        h4_t hv;
        hv[0] = (_Float16)v[0]; hv[1] = (_Float16)v[1];
        hv[2] = (_Float16)v[2]; hv[3] = (_Float16)v[3];
        *(h4_t*)&As[r][c] = hv;
      }
    } else {
      for (int i = tid; i < TILE_M * 32; i += GEMM_BLK) {
        int r = i >> 5, c = i & 31;
        float v = 0.f;
        if (m0 + r < M && k0 + c < K) v = A[(size_t)(m0 + r) * lda + (k0 + c)];
        As[r][c] = (_Float16)v;
      }
    }
    // ---- stage B tile (32 x 64) into Bs[n][k] as f16 ----
    if (fastB && fullK) {
#pragma unroll
      for (int q = 0; q < 4; ++q) {            // 64*32/4 = 512 chunks
        int idx4 = tid + q * GEMM_BLK;
        int n = idx4 >> 3;
        int c = (idx4 & 7) * 4;
        f4_t v;
        if (TRANSB) {
          v = *(const f4_t*)&Bw[(size_t)(n0 + n) * ldb + (k0 + c)];
        } else {
          v[0] = Bw[(size_t)(k0 + c + 0) * ldb + (n0 + n)];
          v[1] = Bw[(size_t)(k0 + c + 1) * ldb + (n0 + n)];
          v[2] = Bw[(size_t)(k0 + c + 2) * ldb + (n0 + n)];
          v[3] = Bw[(size_t)(k0 + c + 3) * ldb + (n0 + n)];
        }
        h4_t hv;
        hv[0] = (_Float16)v[0]; hv[1] = (_Float16)v[1];
        hv[2] = (_Float16)v[2]; hv[3] = (_Float16)v[3];
        *(h4_t*)&Bs[n][c] = hv;
      }
    } else {
      for (int i = tid; i < TILE_N * 32; i += GEMM_BLK) {
        int n = i >> 5, c = i & 31;
        float v = 0.f;
        if (n0 + n < Nc && k0 + c < K)
          v = TRANSB ? Bw[(size_t)(n0 + n) * ldb + (k0 + c)]
                     : Bw[(size_t)(k0 + c) * ldb + (n0 + n)];
        Bs[n][c] = (_Float16)v;
      }
    }
    // prefetch next A K-tile (global_prefetch_b8)
    if (k0 + 32 < K && fastA) {
      int r = tid;
      __builtin_prefetch(&A[(size_t)(m0 + r) * lda + k0 + 32], 0, 3);
    }
    __syncthreads();

    // ---- gather 2 A fragments (16x32 f16 each, ISA 7.12.2 layout) ----
    union { v16h v; h2_t p[8]; } afr[2];
#pragma unroll
    for (int f = 0; f < 2; ++f) {
      const int mr = wave * 32 + f * 16 + ln;
#pragma unroll
      for (int j = 0; j < 8; ++j) {
        int kk = ((j & 4) ? 16 : 0) + hi * 8 + (j & 3) * 2;
        afr[f].p[j] = *(const h2_t*)&As[mr][kk];
      }
    }
    // ---- 4 B fragments; each feeds 2 WMMAs ----
#pragma unroll
    for (int t = 0; t < 4; ++t) {
      union { v16h v; h2_t p[8]; } bfr;
      const int nn = t * 16 + ln;
#pragma unroll
      for (int j = 0; j < 8; ++j) {
        int kk = ((j & 4) ? 16 : 0) + hi * 8 + (j & 3) * 2;
        bfr.p[j] = *(const h2_t*)&Bs[nn][kk];
      }
      acc[0][t] = __builtin_amdgcn_wmma_f32_16x16x32_f16(
          false, afr[0].v, false, bfr.v, (short)0, acc[0][t], false, false);
      acc[1][t] = __builtin_amdgcn_wmma_f32_16x16x32_f16(
          false, afr[1].v, false, bfr.v, (short)0, acc[1][t], false, false);
    }
    __syncthreads();
  }

  // ---- epilogue: C 16x16 f32 layout: M = hi*8 + r, N = lane&15 ----
#pragma unroll
  for (int f = 0; f < 2; ++f) {
#pragma unroll
    for (int t = 0; t < 4; ++t) {
      union { v8f v; float fl[8]; } cu;
      cu.v = acc[f][t];
      const int col = n0 + t * 16 + ln;
      if (col < Nc) {
        const float bv = bias ? bias[col] : 0.f;
#pragma unroll
        for (int r = 0; r < 8; ++r) {
          const int row = m0 + wave * 32 + f * 16 + hi * 8 + r;
          if (row < M) {
            float v = cu.fl[r] + bv;
            if (RESID) v += resid[(size_t)row * ldc + col];
            if (ACT == 1) v = fmaxf(v, 0.f);
            C[(size_t)row * ldc + col] = v;
          }
        }
      }
    }
  }
}

// ---------------------------------------------------------------------------
// Edge gating: e_hat += Dh[src] + Eh[dst];  s = sigmoid(e_hat)
//   num[dst] += s * Bh[src];  den[dst] += s    (segment-sum via f32 atomics)
// ---------------------------------------------------------------------------
__global__ void edge_gate(const float* __restrict__ Dh, const float* __restrict__ Eh,
                          const float* __restrict__ Bh,
                          const int* __restrict__ src, const int* __restrict__ dst,
                          float* __restrict__ ehat,
                          float* __restrict__ num, float* __restrict__ den) {
  const int e = blockIdx.x;
  const int c = threadIdx.x;
  const int s = src[e], d = dst[e];
  const size_t eo = (size_t)e * HDIM + c;
  float v = ehat[eo] + Dh[(size_t)s * HDIM + c] + Eh[(size_t)d * HDIM + c];
  ehat[eo] = v;
  const float sg = 1.f / (1.f + expf(-v));
  atomicAdd(&num[(size_t)d * HDIM + c], sg * Bh[(size_t)s * HDIM + c]);
  atomicAdd(&den[(size_t)d * HDIM + c], sg);
}

__global__ void gcn_combine(const float* __restrict__ Ah, const float* __restrict__ num,
                            const float* __restrict__ den, float* __restrict__ out,
                            long total) {
  long i = (long)blockIdx.x * blockDim.x + threadIdx.x;
  if (i < total) out[i] = Ah[i] + num[i] / (den[i] + 1e-6f);
}

// ---------------------------------------------------------------------------
// BatchNorm (training-mode batch stats, biased variance)
// ---------------------------------------------------------------------------
__global__ void bn_stats(const float* __restrict__ X, long M, int C,
                         float* __restrict__ mean, float* __restrict__ var) {
  const int c = blockIdx.x;
  const int tid = threadIdx.x;
  float s = 0.f, s2 = 0.f;
  for (long i = tid; i < M; i += blockDim.x) {
    float v = X[i * C + c];
    s += v; s2 += v * v;
  }
  __shared__ float sh[256], sh2[256];
  sh[tid] = s; sh2[tid] = s2;
  __syncthreads();
  for (int st = 128; st > 0; st >>= 1) {
    if (tid < st) { sh[tid] += sh[tid + st]; sh2[tid] += sh2[tid + st]; }
    __syncthreads();
  }
  if (tid == 0) {
    float m = sh[0] / (float)M;
    mean[c] = m;
    var[c] = sh2[0] / (float)M - m * m;
  }
}

template <bool RELU, bool RESID>
__global__ void bn_apply(const float* __restrict__ X, const float* __restrict__ resid,
                         float* __restrict__ Y, long total, int C,
                         const float* __restrict__ mean, const float* __restrict__ var,
                         const float* __restrict__ g, const float* __restrict__ b) {
  long i = (long)blockIdx.x * blockDim.x + threadIdx.x;
  if (i >= total) return;
  const int c = (int)(i % C);
  float v = (X[i] - mean[c]) * rsqrtf(var[c] + 1e-5f) * g[c] + b[c];
  if (RELU) v = fmaxf(v, 0.f);
  if (RESID) v += resid[i];
  Y[i] = v;
}

// ---------------------------------------------------------------------------
// Ragged per-graph MHA with online softmax (== masked/padded reference).
// Block per (graph, head). counts[b] = 256 + 8b ; offs[b] = 4b^2 + 252b.
// qkv layout [node][768]: q=0..255, k=256..511, v=512..767.
// ---------------------------------------------------------------------------
__global__ void mha_kernel(const float* __restrict__ qkv, float* __restrict__ out) {
  const int b  = blockIdx.x >> 3;
  const int hh = blockIdx.x & 7;
  const int n  = 256 + 8 * b;
  const int off = 4 * b * b + 252 * b;
  const float scale = 0.17677669529663687f;  // 1/sqrt(32)

  for (int q = threadIdx.x; q < n; q += blockDim.x) {
    const float* qv = qkv + (size_t)(off + q) * 768 + hh * 32;
    float qr[32];
#pragma unroll
    for (int d = 0; d < 32; ++d) qr[d] = qv[d] * scale;

    float mx = -3.0e38f, sum = 0.f, acc[32];
#pragma unroll
    for (int d = 0; d < 32; ++d) acc[d] = 0.f;

    for (int k = 0; k < n; ++k) {
      const float* kv = qkv + (size_t)(off + k) * 768 + 256 + hh * 32;
      const float* vv = kv + 256;
      float s = 0.f;
#pragma unroll
      for (int d = 0; d < 32; ++d) s += qr[d] * kv[d];
      const float nm = fmaxf(mx, s);
      const float corr = expf(mx - nm);
      const float w = expf(s - nm);
      sum = sum * corr + w;
#pragma unroll
      for (int d = 0; d < 32; ++d) acc[d] = acc[d] * corr + w * vv[d];
      mx = nm;
    }
    const float inv = 1.f / sum;
    float* op = out + (size_t)(off + q) * HDIM + hh * 32;
#pragma unroll
    for (int d = 0; d < 32; ++d) op[d] = acc[d] * inv;
  }
}

// ---------------------------------------------------------------------------
// Host-side GEMM dispatch
// ---------------------------------------------------------------------------
static void launch_gemm(const float* A, const float* Bw, const float* bias,
                        const float* resid, float* C, int M, int Nc, int K,
                        int lda, int ldb, int ldc, int act, bool transb,
                        bool residf, hipStream_t stream) {
  dim3 g((M + TILE_M - 1) / TILE_M, (Nc + TILE_N - 1) / TILE_N), blk(GEMM_BLK);
  if (transb)
    gemm_wmma<0, true, false><<<g, blk, 0, stream>>>(A, Bw, bias, resid, C, M, Nc, K, lda, ldb, ldc);
  else if (residf)
    gemm_wmma<0, false, true><<<g, blk, 0, stream>>>(A, Bw, bias, resid, C, M, Nc, K, lda, ldb, ldc);
  else if (act == 1)
    gemm_wmma<1, false, false><<<g, blk, 0, stream>>>(A, Bw, bias, resid, C, M, Nc, K, lda, ldb, ldc);
  else
    gemm_wmma<0, false, false><<<g, blk, 0, stream>>>(A, Bw, bias, resid, C, M, Nc, K, lda, ldb, ldc);
}

static inline long cdiv(long a, long b) { return (a + b - 1) / b; }

// ---------------------------------------------------------------------------
extern "C" void kernel_launch(void* const* d_in, const int* in_sizes, int n_in,
                              void* d_out, int out_size, void* d_ws, size_t ws_size,
                              hipStream_t stream) {
  // ---- input mapping (setup_inputs dict order, params flattened in order) ----
  const float* h_in      = (const float*)d_in[0];   // [N,16]
  const float* e_in      = (const float*)d_in[1];   // [E,16]
  const float* emb_h_w   = (const float*)d_in[2];
  const float* emb_h_b   = (const float*)d_in[3];
  const float* emb_e_w   = (const float*)d_in[4];
  const float* emb_e_b   = (const float*)d_in[5];
  const float* gcn_w     = (const float*)d_in[6];   // [L,5,256,256]
  const float* gcn_b     = (const float*)d_in[7];   // [L,5,256]
  const float* bn_h_g    = (const float*)d_in[8];
  const float* bn_h_b    = (const float*)d_in[9];
  const float* bn_e_g    = (const float*)d_in[10];
  const float* bn_e_b    = (const float*)d_in[11];
  const float* attn_in_w = (const float*)d_in[12];  // [L,768,256]
  const float* attn_in_b = (const float*)d_in[13];  // [L,768]
  const float* attn_ow   = (const float*)d_in[14];  // [L,256,256]
  const float* attn_ob   = (const float*)d_in[15];  // [L,256]
  const float* bn_att_g  = (const float*)d_in[16];
  const float* bn_att_b  = (const float*)d_in[17];
  const float* ffn_w1    = (const float*)d_in[18];  // [256,512]
  const float* ffn_b1    = (const float*)d_in[19];
  const float* ffn_w2    = (const float*)d_in[20];  // [512,256]
  const float* ffn_b2    = (const float*)d_in[21];
  const float* bn_ffn_g  = (const float*)d_in[22];
  const float* bn_ffn_b  = (const float*)d_in[23];
  const float* ro_w1     = (const float*)d_in[24];  // [256,128]
  const float* ro_b1     = (const float*)d_in[25];
  const float* ro_w2     = (const float*)d_in[26];  // [128,64]
  const float* ro_b2     = (const float*)d_in[27];
  const float* ro_w3     = (const float*)d_in[28];  // [64,2]
  const float* ro_b3     = (const float*)d_in[29];
  const int*   esrc      = (const int*)d_in[30];
  const int*   edst      = (const int*)d_in[31];

  const int N = in_sizes[0] / 16;   // 12160
  const int E = in_sizes[30];       // 194560
  const int H = HDIM;

  // ---- workspace carve (floats) ----
  float* p = (float*)d_ws;
  size_t o = 0;
  auto take = [&](size_t n) { float* r = p + o; o += n; return r; };
  float* hbuf = take((size_t)N * H);   // current h
  float* hloc = take((size_t)N * H);   // h_local
  float* Ah   = take((size_t)N * H);
  float* Bh   = take((size_t)N * H);
  float* Dh   = take((size_t)N * H);
  float* Eh   = take((size_t)N * H);
  float* numb = take((size_t)N * H);
  float* denb = take((size_t)N * H);
  float* hg   = take((size_t)N * H);   // gcn pre-BN / attention output scratch
  float* qkvb = take((size_t)N * 768);
  float* midb = take((size_t)N * 512);
  float* r1   = take((size_t)N * 128);
  float* r2   = take((size_t)N * 64);
  float* mnb  = take(768);
  float* vrb  = take(768);
  float* ebuf = take((size_t)E * H);
  float* ehat = take((size_t)E * H);
  (void)ws_size; (void)n_in; (void)out_size;

  const long nh = (long)N * H;
  const long eh = (long)E * H;
  const int T = 256;

  // ---- embeddings ----
  launch_gemm(h_in, emb_h_w, emb_h_b, nullptr, hbuf, N, H, 16, 16, H, H, 0, false, false, stream);
  launch_gemm(e_in, emb_e_w, emb_e_b, nullptr, ebuf, E, H, 16, 16, H, H, 0, false, false, stream);

  for (int i = 0; i < 4; ++i) {
    const float* W  = gcn_w + (size_t)i * 5 * H * H;
    const float* bb = gcn_b + (size_t)i * 5 * H;
    // ---- GatedGCN projections ----
    launch_gemm(hbuf, W + 0 * H * H, bb + 0 * H, nullptr, Ah,  N, H, H, H, H, H, 0, false, false, stream);
    launch_gemm(hbuf, W + 1 * H * H, bb + 1 * H, nullptr, Bh,  N, H, H, H, H, H, 0, false, false, stream);
    launch_gemm(ebuf, W + 2 * H * H, bb + 2 * H, nullptr, ehat, E, H, H, H, H, H, 0, false, false, stream);
    launch_gemm(hbuf, W + 3 * H * H, bb + 3 * H, nullptr, Dh,  N, H, H, H, H, H, 0, false, false, stream);
    launch_gemm(hbuf, W + 4 * H * H, bb + 4 * H, nullptr, Eh,  N, H, H, H, H, H, 0, false, false, stream);

    hipMemsetAsync(numb, 0, (size_t)N * H * sizeof(float), stream);
    hipMemsetAsync(denb, 0, (size_t)N * H * sizeof(float), stream);
    edge_gate<<<E, H, 0, stream>>>(Dh, Eh, Bh, esrc, edst, ehat, numb, denb);
    gcn_combine<<<cdiv(nh, T), T, 0, stream>>>(Ah, numb, denb, hg, nh);

    // h_local = h + relu(BN(h_gcn))    (hbuf kept: attention uses PRE-local h)
    bn_stats<<<H, T, 0, stream>>>(hg, N, H, mnb, vrb);
    bn_apply<true, true><<<cdiv(nh, T), T, 0, stream>>>(hg, hbuf, hloc, nh, H, mnb, vrb,
                                                        bn_h_g + i * H, bn_h_b + i * H);
    // e = e + relu(BN(e_hat))
    bn_stats<<<H, T, 0, stream>>>(ehat, E, H, mnb, vrb);
    bn_apply<true, true><<<cdiv(eh, T), T, 0, stream>>>(ehat, ebuf, ebuf, eh, H, mnb, vrb,
                                                        bn_e_g + i * H, bn_e_b + i * H);

    // ---- global MHA on pre-local h (ragged == masked-padded) ----
    launch_gemm(hbuf, attn_in_w + (size_t)i * 768 * H, attn_in_b + (size_t)i * 768, nullptr,
                qkvb, N, 768, H, H, H, 768, 0, true, false, stream);
    mha_kernel<<<32 * 8, T, 0, stream>>>(qkvb, hg);
    launch_gemm(hg, attn_ow + (size_t)i * H * H, attn_ob + (size_t)i * H, nullptr,
                Ah, N, H, H, H, H, H, 0, true, false, stream);
    // h = h_local + BN(h_global)
    bn_stats<<<H, T, 0, stream>>>(Ah, N, H, mnb, vrb);
    bn_apply<false, true><<<cdiv(nh, T), T, 0, stream>>>(Ah, hloc, hbuf, nh, H, mnb, vrb,
                                                         bn_att_g, bn_att_b);

    // ---- FFN: h = BN(h + W2(relu(W1 h))) ----
    launch_gemm(hbuf, ffn_w1, ffn_b1, nullptr, midb, N, 512, H, H, 512, 512, 1, false, false, stream);
    launch_gemm(midb, ffn_w2, ffn_b2, hbuf, Ah, N, H, 512, 512, H, H, 0, false, true, stream);
    bn_stats<<<H, T, 0, stream>>>(Ah, N, H, mnb, vrb);
    bn_apply<false, false><<<cdiv(nh, T), T, 0, stream>>>(Ah, nullptr, hbuf, nh, H, mnb, vrb,
                                                          bn_ffn_g, bn_ffn_b);
  }

  // ---- MLP readout ----
  launch_gemm(hbuf, ro_w1, ro_b1, nullptr, r1, N, 128, H, H, 128, 128, 1, false, false, stream);
  launch_gemm(r1,   ro_w2, ro_b2, nullptr, r2, N, 64, 128, 128, 64, 64, 1, false, false, stream);
  launch_gemm(r2,   ro_w3, ro_b3, nullptr, (float*)d_out, N, 2, 64, 64, 2, 2, 0, false, false, stream);

  // outputs: (y [N,2], e [E,256]) concatenated flat
  hipMemcpyAsync((float*)d_out + (size_t)N * 2, ebuf, (size_t)E * H * sizeof(float),
                 hipMemcpyDeviceToDevice, stream);
}
